// GroupedQueryAttention_17609365914143
// MI455X (gfx1250) — matrix operand order repaired and checked by reading, not compiled, not run on hardware
//
#include <hip/hip_runtime.h>

typedef __attribute__((ext_vector_type(16))) _Float16 v16h;
typedef __attribute__((ext_vector_type(8)))  _Float16 v8h;
typedef __attribute__((ext_vector_type(8)))  float    v8f;

#define B_SZ    4
#define NTOK    2048
#define DIM_C   1024
#define NHEADS  16
#define KVHEADS 4
#define HDIM    64

static __device__ __forceinline__ v16h combine8(v8h lo, v8h hi) {
  return __builtin_shufflevector(lo, hi, 0,1,2,3,4,5,6,7,8,9,10,11,12,13,14,15);
}

static __device__ __forceinline__ v8f wmma_f16(v16h a, v16h b, v8f c) {
  // emits v_wmma_f32_16x16x32_f16
  return __builtin_amdgcn_wmma_f32_16x16x32_f16(false, a, false, b, (short)0, c, false, false);
}

__global__ void cvt_f32_to_f16(const float* __restrict__ in, _Float16* __restrict__ out, int n) {
  for (int i = blockIdx.x * blockDim.x + threadIdx.x; i < n; i += gridDim.x * blockDim.x)
    out[i] = (_Float16)in[i];
}

// D = A @ W^T  (A: [M,K] f16 row-major, W: [Nout,K] f16 row-major)
// mode 0: store f16 head-major  [B][nh][NTOK][64], value*scale   (Q / K)
// mode 1: store f16 transposed  [B][KVH][64][NTOK]               (V^T)
// mode 2: store f32 [M][1024] + bias                             (output proj)
__global__ __launch_bounds__(256) void gemm_xwT(
    const _Float16* __restrict__ A, const _Float16* __restrict__ W,
    void* __restrict__ dst, const float* __restrict__ bias,
    int Kdim, int mode, int nh, float scale) {
  const int lane = threadIdx.x & 31;
  const int wave = threadIdx.x >> 5;
  const int half = lane >> 4;
  const int ln16 = lane & 15;
  const int m0 = blockIdx.y * 64 + (wave >> 1) * 16;
  const int n0 = blockIdx.x * 128 + (wave & 1) * 64;

  v8f acc[4] = {};
  const _Float16* arow = A + (size_t)(m0 + ln16) * Kdim;
  for (int k0 = 0; k0 < Kdim; k0 += 32) {
    // A-frag 16x32: lane m=l&15; K = idx + 16*half
    v8h alo = *(const v8h*)(arow + k0 + 16 * half);
    v8h ahi = *(const v8h*)(arow + k0 + 16 * half + 8);
    v16h af = combine8(alo, ahi);
#pragma unroll
    for (int c = 0; c < 4; ++c) {
      // B-frag 32x16: lane n=l&15; K = idx + 16*half  (contiguous 32B)
      const _Float16* wrow = W + (size_t)(n0 + c * 16 + ln16) * Kdim + k0 + 16 * half;
      v16h bf = *(const v16h*)wrow;
      acc[c] = wmma_f16(af, bf, acc[c]);
    }
  }
#pragma unroll
  for (int c = 0; c < 4; ++c) {
#pragma unroll
    for (int r = 0; r < 8; ++r) {
      int m = m0 + r + 8 * half;           // D layout: VGPR r -> M=r+8*(l>>4)
      int j = n0 + c * 16 + ln16;          //           N = l&15
      float f = acc[c][r];
      if (mode == 0) {
        int b = m >> 11, tok = m & (NTOK - 1);
        int hh = j >> 6, d = j & 63;
        ((_Float16*)dst)[(((size_t)b * nh + hh) * NTOK + tok) * HDIM + d] = (_Float16)(f * scale);
      } else if (mode == 1) {
        int b = m >> 11, tok = m & (NTOK - 1);
        int kvh = j >> 6, d = j & 63;
        ((_Float16*)dst)[(((size_t)b * KVHEADS + kvh) * HDIM + d) * NTOK + tok] = (_Float16)f;
      } else {
        ((float*)dst)[(size_t)m * DIM_C + j] = f + bias[j];
      }
    }
  }
}

struct alignas(16) WaveLds {
  float    S[16][33];     // 2112 B (multiple of 16 -> P stays 16B aligned)
  _Float16 P[16][40];     // 80 B row stride: 16B-aligned b128 DS loads, conflict-free
  float    alpha[16];
  float    lsum[16];
};

// Flash GQA: 1 wave per 16-query tile; streams 32-key blocks.
__global__ __launch_bounds__(256) void flash_gqa(
    const _Float16* __restrict__ qh, const _Float16* __restrict__ kh,
    const _Float16* __restrict__ vth, _Float16* __restrict__ aoh) {
  __shared__ WaveLds lds[8];
  const int lane = threadIdx.x & 31;
  const int wave = threadIdx.x >> 5;
  const int half = lane >> 4;
  const int ln16 = lane & 15;
  WaveLds& L = lds[wave];

  int t   = blockIdx.x * 8 + wave;   // 8192 tiles total
  int b   = t >> 11;                 // 16 heads * 128 qtiles = 2048 per batch
  int h   = (t >> 7) & 15;
  int qt  = t & 127;
  int kvh = h >> 2;

  const _Float16* qbase = qh  + (((size_t)b * NHEADS + h) * NTOK + qt * 16) * HDIM;
  const _Float16* kbase = kh  + ((size_t)b * KVHEADS + kvh) * NTOK * HDIM;
  const _Float16* vbase = vth + ((size_t)b * KVHEADS + kvh) * HDIM * NTOK;

  // Q A-fragments for kd 0..31 and 32..63 (scale pre-folded into qh)
  const _Float16* qrow = qbase + ln16 * HDIM;
  v16h qa0 = combine8(*(const v8h*)(qrow +      16 * half), *(const v8h*)(qrow +  8 + 16 * half));
  v16h qa1 = combine8(*(const v8h*)(qrow + 32 + 16 * half), *(const v8h*)(qrow + 40 + 16 * half));

  v8f of[4] = {};
  float m_run = -3.0e38f, l_run = 0.0f;
  const int row = ln16;

  for (int blk = 0; blk < NTOK / 32; ++blk) {
    const int key0 = blk * 32;
    // ---- S = Q @ K^T  (two 16-key subtiles, K=64 chained) ----
#pragma unroll
    for (int sub = 0; sub < 2; ++sub) {
      const _Float16* krow = kbase + (size_t)(key0 + sub * 16 + ln16) * HDIM + 16 * half;
      v16h kb0 = *(const v16h*)(krow);
      v16h kb1 = *(const v16h*)(krow + 32);
      v8f s = {};
      s = wmma_f16(qa0, kb0, s);
      s = wmma_f16(qa1, kb1, s);
#pragma unroll
      for (int r = 0; r < 8; ++r)
        L.S[r + 8 * half][sub * 16 + ln16] = s[r];
    }
    __syncthreads();
    // ---- streaming softmax: lane handles row=l&15, cols [16*half,16*half+16) ----
    float pm = -3.0e38f;
#pragma unroll
    for (int j = 0; j < 16; ++j) pm = fmaxf(pm, L.S[row][16 * half + j]);
    pm = fmaxf(pm, __shfl_xor(pm, 16));
    float m_new = fmaxf(m_run, pm);
    float ps = 0.0f;
#pragma unroll
    for (int j = 0; j < 16; ++j) {
      float p = __expf(L.S[row][16 * half + j] - m_new);
      L.P[row][16 * half + j] = (_Float16)p;
      ps += p;
    }
    ps += __shfl_xor(ps, 16);
    float al = __expf(m_run - m_new);
    l_run = l_run * al + ps;
    m_run = m_new;
    if (lane < 16) L.alpha[row] = al;
    __syncthreads();
    // ---- rescale O, then O += P @ V ----
    v16h pa = combine8(*(const v8h*)&L.P[ln16][16 * half],
                       *(const v8h*)&L.P[ln16][16 * half + 8]);
#pragma unroll
    for (int r = 0; r < 8; ++r) {
      float a_r = L.alpha[r + 8 * half];
#pragma unroll
      for (int c = 0; c < 4; ++c) of[c][r] *= a_r;
    }
#pragma unroll
    for (int c = 0; c < 4; ++c) {
      const _Float16* vrow = vbase + (size_t)(c * 16 + ln16) * NTOK + key0 + 16 * half;
      v16h vb = *(const v16h*)vrow;
      of[c] = wmma_f16(pa, vb, of[c]);
    }
  }

  if (lane < 16) L.lsum[row] = l_run;
  __syncthreads();
#pragma unroll
  for (int r = 0; r < 8; ++r) {
    float inv = 1.0f / L.lsum[r + 8 * half];
    int tok = qt * 16 + r + 8 * half;
    size_t base = ((size_t)b * NTOK + tok) * DIM_C + h * HDIM;
#pragma unroll
    for (int c = 0; c < 4; ++c)
      aoh[base + c * 16 + ln16] = (_Float16)(of[c][r] * inv);
  }
}

extern "C" void kernel_launch(void* const* d_in, const int* in_sizes, int n_in,
                              void* d_out, int out_size, void* d_ws, size_t ws_size,
                              hipStream_t stream) {
  (void)in_sizes; (void)n_in; (void)out_size; (void)ws_size;
  const float* x  = (const float*)d_in[0];
  const float* wq = (const float*)d_in[1];
  const float* wk = (const float*)d_in[2];
  const float* wv = (const float*)d_in[3];
  const float* wo = (const float*)d_in[4];
  const float* bo = (const float*)d_in[5];

  char* ws = (char*)d_ws;
  size_t off = 0;
  auto alloc = [&](size_t bytes) { char* p = ws + off; off += (bytes + 255) & ~(size_t)255; return p; };
  _Float16* xh  = (_Float16*)alloc((size_t)8192 * 1024 * 2);
  _Float16* wqh = (_Float16*)alloc((size_t)1024 * 1024 * 2);
  _Float16* wkh = (_Float16*)alloc((size_t)256  * 1024 * 2);
  _Float16* wvh = (_Float16*)alloc((size_t)256  * 1024 * 2);
  _Float16* woh = (_Float16*)alloc((size_t)1024 * 1024 * 2);
  _Float16* qhb = (_Float16*)alloc((size_t)8192 * 1024 * 2);  // [B][16][N][64]
  _Float16* khb = (_Float16*)alloc((size_t)8192 * 256  * 2);  // [B][4][N][64]
  _Float16* vtb = (_Float16*)alloc((size_t)8192 * 256  * 2);  // [B][4][64][N]
  _Float16* aoh = (_Float16*)alloc((size_t)8192 * 1024 * 2);  // [B][N][C]

  cvt_f32_to_f16<<<2048, 256, 0, stream>>>(x,  xh,  8192 * 1024);
  cvt_f32_to_f16<<<1024, 256, 0, stream>>>(wq, wqh, 1024 * 1024);
  cvt_f32_to_f16<<<256,  256, 0, stream>>>(wk, wkh, 256 * 1024);
  cvt_f32_to_f16<<<256,  256, 0, stream>>>(wv, wvh, 256 * 1024);
  cvt_f32_to_f16<<<1024, 256, 0, stream>>>(wo, woh, 1024 * 1024);

  const float qscale = 0.125f;  // HEAD_DIM^-0.5
  gemm_xwT<<<dim3(8, 128), 256, 0, stream>>>(xh, wqh, qhb, nullptr, 1024, 0, 16, qscale);
  gemm_xwT<<<dim3(2, 128), 256, 0, stream>>>(xh, wkh, khb, nullptr, 1024, 0, 4, 1.0f);
  gemm_xwT<<<dim3(2, 128), 256, 0, stream>>>(xh, wvh, vtb, nullptr, 1024, 1, 4, 1.0f);

  flash_gqa<<<1024, 256, 0, stream>>>(qhb, khb, vtb, aoh);

  gemm_xwT<<<dim3(8, 128), 256, 0, stream>>>(aoh, woh, d_out, bo, 1024, 2, 0, 1.0f);
}
